// Transformer_40604620817017
// MI455X (gfx1250) — matrix operand order repaired and checked by reading, not compiled, not run on hardware
//
#include <hip/hip_runtime.h>
#include <hip/hip_bf16.h>

typedef __attribute__((ext_vector_type(16))) _Float16 v16h;
typedef __attribute__((ext_vector_type(8)))  _Float16 v8h;
typedef __attribute__((ext_vector_type(4)))  _Float16 v4h;
typedef __attribute__((ext_vector_type(8)))  float    v8f;

// Model constants
constexpr int CV   = 32000;
constexpr int CD   = 512;
constexpr int CH   = 8;
constexpr int CHD  = 64;     // CD / CH
constexpr int CL   = 6;
constexpr int CDFF = 2048;
constexpr int CB   = 64;
constexpr int CT   = 256;
constexpr int BT   = CB * CT;   // 16384
constexpr int PAD_IDX = 1;

// ---------------------------------------------------------------------------
// Generic batched GEMM:  C = act(alpha * A @ B + bias [+ residual])
//   A element (m,k)  : A + batch_offs + m*lda + k            (K-contiguous)
//   B element (k,n)  : B + batch_offs + k*ldbk + n*ldbn      (either contiguity)
//   C element (m,n)  : C + batch_offs + m*ldc + n
//
// Block: 256 threads = 8 waves. Block tile 64(M) x 64(N), K step 32.
// A tile staged row-major [64 x 32] f16 (stride 40 halfs, 16B-aligned rows);
// B tile staged TRANSPOSED [n=64 x k=32] f16 (stride 40) so that each wave's
// WMMA fragments are contiguous 16B runs -> ds_load_b128.
// Each wave owns one 16-row M sub-tile and two 16-col N sub-tiles:
// one A fragment feeds two v_wmma_f32_16x16x32_f16 per K step.
// ---------------------------------------------------------------------------
#define TS 40   // LDS tile stride in halfs (80 B: multiple of 16 B, padded)

__global__ __launch_bounds__(256) void gemm_f16(
    const float* __restrict__ A, const float* __restrict__ Bm,
    const float* __restrict__ bias, const float* __restrict__ Rsd,
    float* __restrict__ C, int K,
    int lda, int ldbk, int ldbn, int ldc,
    long long sAb, long long sAh, long long sBb, long long sBh,
    long long sCb, long long sCh, int Hdim, float alpha, int relu)
{
    __shared__ __align__(16) _Float16 As[64 * TS];  // [m][k]
    __shared__ __align__(16) _Float16 Bt[64 * TS];  // [n][k]  (transposed)

    const int bz = blockIdx.z;
    const int bo = bz / Hdim;
    const int ho = bz % Hdim;
    const float* Ab = A  + (long long)bo * sAb + (long long)ho * sAh;
    const float* Bb = Bm + (long long)bo * sBb + (long long)ho * sBh;
    float*       Cb = C  + (long long)bo * sCb + (long long)ho * sCh;
    const float* Rb = Rsd ? (Rsd + (long long)bo * sCb + (long long)ho * sCh) : nullptr;

    const int m0   = blockIdx.y * 64;
    const int n0   = blockIdx.x * 64;
    const int tid  = threadIdx.x;
    const int lane = tid & 31;
    const int wid  = tid >> 5;
    const int mt   = wid & 3;         // M sub-tile 0..3
    const int np   = wid >> 2;        // N pair 0..1 -> n sub-tiles 2*np, 2*np+1
    const int nlan = lane & 15;
    const int grp  = lane >> 4;

    v8f acc0 = {};
    v8f acc1 = {};

    for (int k0 = 0; k0 < K; k0 += 32) {
        // ---- stage A tile: 64 rows x 32 K, fp32 -> f16, float4 loads ----
        #pragma unroll
        for (int i = 0; i < 2; ++i) {
            int e = tid + i * 256;          // 0..511
            int r = e >> 3;                 // 0..63
            int c = (e & 7) * 4;            // 0,4,..,28
            const float* src = Ab + (long long)(m0 + r) * lda + (k0 + c);
            if (k0 + 32 < K) __builtin_prefetch(src + 32, 0, 1);
            float4 f = *(const float4*)src;
            v4h h = { (_Float16)f.x, (_Float16)f.y, (_Float16)f.z, (_Float16)f.w };
            *(v4h*)&As[r * TS + c] = h;
        }
        // ---- stage B tile transposed: Bt[n][k], 64 n x 32 k ----
        if (ldbk == 1) {
            // B is K-contiguous (attention Q@K^T): vector read along k
            #pragma unroll
            for (int i = 0; i < 2; ++i) {
                int e  = tid + i * 256;
                int nn = e >> 3;            // 0..63
                int kk = (e & 7) * 4;       // 0..28
                const float* src = Bb + (long long)(n0 + nn) * ldbn + (k0 + kk);
                if (k0 + 32 < K) __builtin_prefetch(src + 32, 0, 1);
                float4 f = *(const float4*)src;
                v4h h = { (_Float16)f.x, (_Float16)f.y, (_Float16)f.z, (_Float16)f.w };
                *(v4h*)&Bt[nn * TS + kk] = h;
            }
        } else {
            // B is N-contiguous (weights, wei@V): vector read along n, scatter halfs
            #pragma unroll
            for (int i = 0; i < 2; ++i) {
                int e  = tid + i * 256;
                int kk = e >> 4;            // 0..31
                int nn = (e & 15) * 4;      // 0..60
                const float* src = Bb + (long long)(k0 + kk) * ldbk + (n0 + nn);
                if (k0 + 32 < K) __builtin_prefetch(src + (long long)32 * ldbk, 0, 1);
                float4 f = *(const float4*)src;
                Bt[(nn + 0) * TS + kk] = (_Float16)f.x;
                Bt[(nn + 1) * TS + kk] = (_Float16)f.y;
                Bt[(nn + 2) * TS + kk] = (_Float16)f.z;
                Bt[(nn + 3) * TS + kk] = (_Float16)f.w;
            }
        }
        __syncthreads();

        // ---- fragments: contiguous 16B runs -> ds_load_b128 ----
        // A (16x32, M per lane): af[0..7]=K grp*8..+7, af[8..15]=K 16+grp*8..+7
        const _Float16* ar = &As[(mt * 16 + nlan) * TS];
        v8h a0 = *(const v8h*)(ar + grp * 8);
        v8h a1 = *(const v8h*)(ar + 16 + grp * 8);
        v16h af = __builtin_shufflevector(a0, a1,
            0,1,2,3,4,5,6,7, 8,9,10,11,12,13,14,15);
        // B (32x16, N per lane): bf[0..7]=K grp*8..+7, bf[8..15]=K 16+grp*8..+7 for column ncol
        const _Float16* br0 = &Bt[(np * 32 + nlan) * TS + grp * 8];
        const _Float16* br1 = br0 + 16 * TS;
        v8h b00 = *(const v8h*)br0;
        v8h b01 = *(const v8h*)(br0 + 16);
        v8h b10 = *(const v8h*)br1;
        v8h b11 = *(const v8h*)(br1 + 16);
        v16h bf0 = __builtin_shufflevector(b00, b01,
            0,1,2,3,4,5,6,7, 8,9,10,11,12,13,14,15);
        v16h bf1 = __builtin_shufflevector(b10, b11,
            0,1,2,3,4,5,6,7, 8,9,10,11,12,13,14,15);

        acc0 = __builtin_amdgcn_wmma_f32_16x16x32_f16(
            false, af, false, bf0, (short)0, acc0, false, false);
        acc1 = __builtin_amdgcn_wmma_f32_16x16x32_f16(
            false, af, false, bf1, (short)0, acc1, false, false);
        __syncthreads();
    }

    // ---- epilogue: alpha, bias, residual, relu ----
    #pragma unroll
    for (int t = 0; t < 2; ++t) {
        const v8f acc = t ? acc1 : acc0;
        const int n = n0 + (np * 2 + t) * 16 + nlan;
        const float bv = bias ? bias[n] : 0.0f;
        #pragma unroll
        for (int r = 0; r < 8; ++r) {
            int m = m0 + mt * 16 + grp * 8 + r;
            float val = acc[r] * alpha + bv;
            if (Rb) val += Rb[(long long)m * ldc + n];
            if (relu) val = val > 0.0f ? val : 0.0f;
            Cb[(long long)m * ldc + n] = val;
        }
    }
}

// ---------------------------------------------------------------------------
// Embedding + sinusoidal positional encoding. One block per (b,t) row, D=512.
// ---------------------------------------------------------------------------
__global__ __launch_bounds__(256) void embed_pe(
    const int* __restrict__ idx, const float* __restrict__ emb,
    float* __restrict__ out)
{
    long long bt = blockIdx.x;
    int t   = (int)(bt % CT);
    int tok = idx[bt];
    const float* e = emb + (long long)tok * CD;
    float* o = out + bt * CD;
    int j = threadIdx.x;
    #pragma unroll
    for (int s = 0; s < 2; ++s) {
        int d = j + s * 256;
        int pair = d >> 1;
        float ang = (float)t * __powf(10000.0f, -(float)(2 * pair) / (float)CD);
        float pe = (d & 1) ? __cosf(ang) : __sinf(ang);
        o[d] = e[d] + pe;
    }
}

// ---------------------------------------------------------------------------
// LayerNorm: one block per row, D=512 (2 elements per thread).
// ---------------------------------------------------------------------------
__global__ __launch_bounds__(256) void layernorm_k(
    const float* __restrict__ x, const float* __restrict__ g,
    const float* __restrict__ b, float* __restrict__ y)
{
    long long row = blockIdx.x;
    const float* xr = x + row * CD;
    float* yr = y + row * CD;
    int t = threadIdx.x;
    float a0 = xr[t], a1 = xr[t + 256];
    __shared__ float r1[256], r2[256];
    r1[t] = a0 + a1;
    r2[t] = a0 * a0 + a1 * a1;
    __syncthreads();
    for (int st = 128; st > 0; st >>= 1) {
        if (t < st) { r1[t] += r1[t + st]; r2[t] += r2[t + st]; }
        __syncthreads();
    }
    float mu  = r1[0] * (1.0f / CD);
    float var = r2[0] * (1.0f / CD) - mu * mu;
    float rs  = rsqrtf(var + 1e-5f);
    yr[t]       = (a0 - mu) * rs * g[t]       + b[t];
    yr[t + 256] = (a1 - mu) * rs * g[t + 256] + b[t + 256];
}

// ---------------------------------------------------------------------------
// Row softmax over T=256 with optional causal mask. One block per row.
// ---------------------------------------------------------------------------
__global__ __launch_bounds__(256) void softmax_rows(
    float* __restrict__ s, int causal)
{
    long long row = blockIdx.x;
    int qi = (int)(row % CT);
    float* p = s + row * CT;
    int j = threadIdx.x;
    int len = causal ? (qi + 1) : CT;
    float x = (j < len) ? p[j] : -INFINITY;
    __shared__ float red[256];
    red[j] = x; __syncthreads();
    for (int st = 128; st > 0; st >>= 1) {
        if (j < st) red[j] = fmaxf(red[j], red[j + st]);
        __syncthreads();
    }
    float mx = red[0];
    __syncthreads();
    float e = (j < len) ? __expf(x - mx) : 0.0f;
    red[j] = e; __syncthreads();
    for (int st = 128; st > 0; st >>= 1) {
        if (j < st) red[j] += red[j + st];
        __syncthreads();
    }
    p[j] = e / red[0];
}

// ---------------------------------------------------------------------------
// Cross entropy: one block per row of logits [BT, V]; masked accumulate.
// ---------------------------------------------------------------------------
__global__ __launch_bounds__(256) void ce_row(
    const float* __restrict__ logits, const int* __restrict__ tgt,
    float* __restrict__ accum)
{
    long long row = blockIdx.x;
    const float* p = logits + row * (long long)CV;
    int t = threadIdx.x;
    float mx = -INFINITY;
    for (int j = t; j < CV; j += 256) mx = fmaxf(mx, p[j]);
    __shared__ float red[256];
    red[t] = mx; __syncthreads();
    for (int st = 128; st > 0; st >>= 1) {
        if (t < st) red[t] = fmaxf(red[t], red[t + st]);
        __syncthreads();
    }
    mx = red[0];
    __syncthreads();
    float s = 0.0f;
    for (int j = t; j < CV; j += 256) s += __expf(p[j] - mx);
    red[t] = s; __syncthreads();
    for (int st = 128; st > 0; st >>= 1) {
        if (t < st) red[t] += red[t + st];
        __syncthreads();
    }
    if (t == 0) {
        int tv = tgt[row];
        if (tv != PAD_IDX) {
            float lse = mx + __logf(red[0]);
            atomicAdd(&accum[0], lse - p[tv]);
            atomicAdd(&accum[1], 1.0f);
        }
    }
}

__global__ void zero2(float* a)
{
    if (threadIdx.x < 2) a[threadIdx.x] = 0.0f;
}

__global__ void finalize_loss(const float* __restrict__ accum, float* __restrict__ out)
{
    if (threadIdx.x == 0) out[0] = accum[0] / fmaxf(accum[1], 1.0f);
}

// ---------------------------------------------------------------------------
// Host-side launch helpers
// ---------------------------------------------------------------------------
static inline void launch_gemm(hipStream_t st,
    const float* A, const float* Bm, const float* bias, const float* res,
    float* C, int M, int N, int K,
    int lda, int ldbk, int ldbn, int ldc,
    long long sAb, long long sAh, long long sBb, long long sBh,
    long long sCb, long long sCh, int Hdim, int batches,
    float alpha, int relu)
{
    dim3 grid(N / 64, M / 64, batches);
    hipLaunchKernelGGL(gemm_f16, grid, dim3(256), 0, st,
        A, Bm, bias, res, C, K, lda, ldbk, ldbn, ldc,
        sAb, sAh, sBb, sBh, sCb, sCh, Hdim, alpha, relu);
}

static void mha(hipStream_t st,
    const float* hq, const float* hkv,
    const float* Wq, const float* Wk, const float* Wv,
    const float* Wo, const float* bo,
    float* resid_io, float* q, float* k, float* v, float* att, float* scores,
    int causal)
{
    const long long sTD = (long long)CT * CD;
    const long long sTT = (long long)CT * CT;
    // projections: [BT,D] @ [D,D]
    launch_gemm(st, hq,  Wq, nullptr, nullptr, q, BT, CD, CD, CD, CD, 1, CD,
                0,0,0,0,0,0, 1, 1, 1.0f, 0);
    launch_gemm(st, hkv, Wk, nullptr, nullptr, k, BT, CD, CD, CD, CD, 1, CD,
                0,0,0,0,0,0, 1, 1, 1.0f, 0);
    launch_gemm(st, hkv, Wv, nullptr, nullptr, v, BT, CD, CD, CD, CD, 1, CD,
                0,0,0,0,0,0, 1, 1, 1.0f, 0);
    // scores[b,h] = (q[b,:,h,:] @ k[b,:,h,:]^T) * hd^-0.5   (batched 256x256x64)
    launch_gemm(st, q, k, nullptr, nullptr, scores, CT, CT, CHD,
                CD, /*ldbk=*/1, /*ldbn=*/CD, /*ldc=*/CT,
                sTD, CHD, sTD, CHD, (long long)CH * sTT, sTT,
                CH, CB * CH, 0.125f, 0);
    hipLaunchKernelGGL(softmax_rows, dim3(CB * CH * CT), dim3(256), 0, st,
                       scores, causal);
    // att[b,:,h,:] = wei[b,h] @ v[b,:,h,:]   (batched 256x64x256)
    launch_gemm(st, scores, v, nullptr, nullptr, att, CT, CHD, CT,
                CT, /*ldbk=*/CD, /*ldbn=*/1, /*ldc=*/CD,
                (long long)CH * sTT, sTT, sTD, CHD, sTD, CHD,
                CH, CB * CH, 1.0f, 0);
    // resid_io += att @ Wo + bo
    launch_gemm(st, att, Wo, bo, resid_io, resid_io, BT, CD, CD,
                CD, CD, 1, CD, 0,0,0,0,0,0, 1, 1, 1.0f, 0);
}

static void ffn(hipStream_t st, const float* h,
    const float* W1, const float* b1, const float* W2, const float* b2,
    float* resid_io, float* hid)
{
    launch_gemm(st, h, W1, b1, nullptr, hid, BT, CDFF, CD,
                CD, CDFF, 1, CDFF, 0,0,0,0,0,0, 1, 1, 1.0f, 1 /*relu*/);
    launch_gemm(st, hid, W2, b2, resid_io, resid_io, BT, CD, CDFF,
                CDFF, CD, 1, CD, 0,0,0,0,0,0, 1, 1, 1.0f, 0);
}

extern "C" void kernel_launch(void* const* d_in, const int* in_sizes, int n_in,
                              void* d_out, int out_size, void* d_ws, size_t ws_size,
                              hipStream_t stream)
{
    const int*   idx      = (const int*)  d_in[0];
    const int*   idx_enc  = (const int*)  d_in[1];
    const int*   targets  = (const int*)  d_in[2];
    const float* emb_dec  = (const float*)d_in[3];
    const float* emb_enc  = (const float*)d_in[4];
    const float* enc_Wq   = (const float*)d_in[5];
    const float* enc_Wk   = (const float*)d_in[6];
    const float* enc_Wv   = (const float*)d_in[7];
    const float* enc_Wo   = (const float*)d_in[8];
    const float* enc_bo   = (const float*)d_in[9];
    const float* enc_ln_g = (const float*)d_in[10];
    const float* enc_ln_b = (const float*)d_in[11];
    const float* enc_W1   = (const float*)d_in[12];
    const float* enc_b1   = (const float*)d_in[13];
    const float* enc_W2   = (const float*)d_in[14];
    const float* enc_b2   = (const float*)d_in[15];
    const float* dWq_sa   = (const float*)d_in[16];
    const float* dWk_sa   = (const float*)d_in[17];
    const float* dWv_sa   = (const float*)d_in[18];
    const float* dWo_sa   = (const float*)d_in[19];
    const float* dbo_sa   = (const float*)d_in[20];
    const float* dWq_xa   = (const float*)d_in[21];
    const float* dWk_xa   = (const float*)d_in[22];
    const float* dWv_xa   = (const float*)d_in[23];
    const float* dWo_xa   = (const float*)d_in[24];
    const float* dbo_xa   = (const float*)d_in[25];
    const float* dec_ln_g = (const float*)d_in[26];
    const float* dec_ln_b = (const float*)d_in[27];
    const float* dec_W1   = (const float*)d_in[28];
    const float* dec_b1   = (const float*)d_in[29];
    const float* dec_W2   = (const float*)d_in[30];
    const float* dec_b2   = (const float*)d_in[31];
    const float* lnf_g    = (const float*)d_in[32];
    const float* lnf_b    = (const float*)d_in[33];
    const float* Wlm      = (const float*)d_in[34];
    const float* blm      = (const float*)d_in[35];
    (void)in_sizes; (void)n_in; (void)out_size; (void)ws_size;

    float* out = (float*)d_out;   // [BT*V] logits, then 1 loss scalar

    float* ws = (float*)d_ws;
    const long long nTD  = (long long)BT * CD;
    const long long nFFN = (long long)BT * CDFF;
    const long long nSC  = (long long)CB * CH * CT * CT;
    float* y      = ws;
    float* x      = y   + nTD;
    float* h      = x   + nTD;
    float* hk     = h   + nTD;
    float* q      = hk  + nTD;
    float* kbuf   = q   + nTD;
    float* vbuf   = kbuf+ nTD;
    float* att    = vbuf+ nTD;
    float* ffnbuf = att + nTD;
    float* scores = ffnbuf + nFFN;
    float* accum  = scores + nSC;

    const long long LDD  = (long long)CD * CD;
    const long long LDF  = (long long)CD * CDFF;

    // ---------------- Encoder ----------------
    hipLaunchKernelGGL(embed_pe, dim3(BT), dim3(256), 0, stream, idx_enc, emb_enc, y);
    for (int l = 0; l < CL; ++l) {
        const float* lg = enc_ln_g + (long long)l * 2 * CD;
        const float* lb = enc_ln_b + (long long)l * 2 * CD;
        hipLaunchKernelGGL(layernorm_k, dim3(BT), dim3(256), 0, stream, y, lg, lb, h);
        mha(stream, h, h,
            enc_Wq + l * LDD, enc_Wk + l * LDD, enc_Wv + l * LDD,
            enc_Wo + l * LDD, enc_bo + (long long)l * CD,
            y, q, kbuf, vbuf, att, scores, /*causal=*/0);
        hipLaunchKernelGGL(layernorm_k, dim3(BT), dim3(256), 0, stream, y, lg + CD, lb + CD, h);
        ffn(stream, h, enc_W1 + l * LDF, enc_b1 + (long long)l * CDFF,
            enc_W2 + l * LDF, enc_b2 + (long long)l * CD, y, ffnbuf);
    }

    // ---------------- Decoder ----------------
    hipLaunchKernelGGL(embed_pe, dim3(BT), dim3(256), 0, stream, idx, emb_dec, x);
    for (int l = 0; l < CL; ++l) {
        const float* lg = dec_ln_g + (long long)l * 4 * CD;
        const float* lb = dec_ln_b + (long long)l * 4 * CD;
        hipLaunchKernelGGL(layernorm_k, dim3(BT), dim3(256), 0, stream, x, lg, lb, h);
        mha(stream, h, h,
            dWq_sa + l * LDD, dWk_sa + l * LDD, dWv_sa + l * LDD,
            dWo_sa + l * LDD, dbo_sa + (long long)l * CD,
            x, q, kbuf, vbuf, att, scores, /*causal=*/1);
        hipLaunchKernelGGL(layernorm_k, dim3(BT), dim3(256), 0, stream, x, lg + CD, lb + CD, h);
        hipLaunchKernelGGL(layernorm_k, dim3(BT), dim3(256), 0, stream, y, lg + 2 * CD, lb + 2 * CD, hk);
        mha(stream, h, hk,
            dWq_xa + l * LDD, dWk_xa + l * LDD, dWv_xa + l * LDD,
            dWo_xa + l * LDD, dbo_xa + (long long)l * CD,
            x, q, kbuf, vbuf, att, scores, /*causal=*/0);
        hipLaunchKernelGGL(layernorm_k, dim3(BT), dim3(256), 0, stream, x, lg + 3 * CD, lb + 3 * CD, h);
        ffn(stream, h, dec_W1 + l * LDF, dec_b1 + (long long)l * CDFF,
            dec_W2 + l * LDF, dec_b2 + (long long)l * CD, x, ffnbuf);
    }

    // ---------------- LM head + loss ----------------
    hipLaunchKernelGGL(layernorm_k, dim3(BT), dim3(256), 0, stream, x, lnf_g, lnf_b, h);
    launch_gemm(stream, h, Wlm, blm, nullptr, out, BT, CV, CD,
                CD, /*ldbk=*/CV, /*ldbn=*/1, /*ldc=*/CV,
                0,0,0,0,0,0, 1, 1, 1.0f, 0);
    hipLaunchKernelGGL(zero2, dim3(1), dim3(32), 0, stream, accum);
    hipLaunchKernelGGL(ce_row, dim3(BT), dim3(256), 0, stream, out, targets, accum);
    hipLaunchKernelGGL(finalize_loss, dim3(1), dim3(32), 0, stream,
                       accum, out + (long long)BT * CV);
}
